// TorchAttention_71347996721677
// MI455X (gfx1250) — matrix operand order repaired and checked by reading, not compiled, not run on hardware
//
#include <hip/hip_runtime.h>

typedef __attribute__((ext_vector_type(16))) _Float16 v16h;
typedef __attribute__((ext_vector_type(8)))  _Float16 v8h;
typedef __attribute__((ext_vector_type(8)))  float    v8f;
typedef __attribute__((ext_vector_type(4)))  float    v4f;

#define BATCH  64
#define SEQ    256
#define DMODEL 2048
#define NHEAD  16
#define HDIM   128

#define TM 128
#define TN 128
#define TK 32
#define NIT (DMODEL / TK)   // 64 K-chunks
#define LDH 40              // LDS row stride in halves (80B: 16B-aligned, conflict-free)
#define KLDH 136            // K-tile LDS row stride in halves (272B)

// ---- Tensor Data Mover (ROCm 7.2 / clang-22 5-arg builtin); else async fallback
#if defined(__clang_major__) && (__clang_major__ <= 22) && __has_builtin(__builtin_amdgcn_tensor_load_to_lds)
#define HAVE_TDM 1
#else
#define HAVE_TDM 0
#endif

#if HAVE_TDM
typedef __attribute__((ext_vector_type(4))) unsigned int u32x4;
typedef __attribute__((ext_vector_type(8))) int i32x8;
typedef __attribute__((ext_vector_type(4))) int i32x4;

// 2D f16 tile load via TDM.  Dims/strides in elements; pad encodings per D# spec:
// pad_interval enc: 0=2,1=4,2=8,3=16,...,5=64 DWORDs (set == row length for per-row pad)
// pad_amount enc:   n -> n+1 DWORDs
static __device__ __forceinline__ void tdm_load_2d(
    unsigned lds_addr, unsigned long long gaddr,
    unsigned tile_d0, unsigned tile_d1,
    unsigned tensor_d0, unsigned tensor_d1, unsigned stride_d0,
    unsigned pad_int, unsigned pad_amt)
{
  u32x4 g0;
  g0[0] = 1u;                                                  // count=1, user mode
  g0[1] = lds_addr;                                            // lds_addr [63:32]
  g0[2] = (unsigned)(gaddr & 0xFFFFFFFFull);                   // global_addr lo
  g0[3] = (unsigned)((gaddr >> 32) & 0x1FFFFFFull) | (2u << 30); // addr[56:32] | type=2
  i32x8 g1;
  g1[0] = (int)((1u << 16) | (1u << 20) | (pad_int << 22) | (pad_amt << 25)); // data_size=2B, pad_enable
  g1[1] = (int)((tensor_d0 & 0xFFFFu) << 16);                  // dim0[15:0] @ [63:48]
  g1[2] = (int)(((tensor_d0 >> 16) & 0xFFFFu) | ((tensor_d1 & 0xFFFFu) << 16));
  g1[3] = (int)(((tensor_d1 >> 16) & 0xFFFFu) | ((tile_d0 & 0xFFFFu) << 16));
  g1[4] = (int)(tile_d1 & 0xFFFFu);                            // tile_dim1; tile_dim2=0
  g1[5] = (int)stride_d0;                                      // dim0_stride[31:0]
  g1[6] = 0;                                                   // stride hi / dim1_stride lo
  g1[7] = 0;
  i32x4 gz = {0, 0, 0, 0};
  __builtin_amdgcn_tensor_load_to_lds(g0, g1, gz, gz, 0);
}
#endif

static __device__ __forceinline__ v16h cat16(v8h lo, v8h hi) {
  v16h r;
#pragma unroll
  for (int i = 0; i < 8; ++i) { r[i] = lo[i]; r[i + 8] = hi[i]; }
  return r;
}

// 32B global -> LDS async copy (ASYNCcnt); inst offset applies to both addresses.
static __device__ __forceinline__ void async_copy32(unsigned lds, unsigned long long ga) {
  asm volatile("global_load_async_to_lds_b128 %0, %1, off\n\t"
               "global_load_async_to_lds_b128 %0, %1, off offset:16"
               :: "v"(lds), "v"(ga) : "memory");
}

// ---------------------------------------------------------------------------
// Y = (X @ Wh^T + bias) * scale.  Wh: N x K f16 (pre-converted).
// XF16=0: X fp32 (overlapped convert staging); XF16=1: X f16 (TDM/async).
// MODE 0: f16 (B,H,S,HD) | 1: f16 (B,H,HD,S) | 2: fp32 MxN
// ---------------------------------------------------------------------------
template <int XF16, int MODE>
__global__ __launch_bounds__(256)
void gemm_xwT(const void* __restrict__ Xv, const _Float16* __restrict__ Wh,
              const float* __restrict__ bias, void* __restrict__ Yv, float scale)
{
  __shared__ __attribute__((aligned(16))) _Float16 Xs[2][TM * LDH];
  __shared__ __attribute__((aligned(16))) _Float16 Ws[2][TN * LDH];

  const int K = DMODEL, N = DMODEL, M = 16384;
  const int tid  = threadIdx.x;
  const int lane = tid & 31;
  const int w    = tid >> 5;
  const int wm   = w & 3;          // 4 waves x 32 rows
  const int wn   = w >> 2;         // 2 waves x 64 cols
  const int m0 = blockIdx.x * TM;
  const int n0 = blockIdx.y * TN;

  const float*    Xf = (const float*)Xv;
  const _Float16* Xh = (const _Float16*)Xv;

  const int srow = tid >> 1;
  const int scol = (tid & 1) * 16;
  const size_t   wgoff  = (size_t)(n0 + srow) * K + scol;
  const size_t   xgoff  = (size_t)(m0 + srow) * K + scol;
  const unsigned ldsoff = (unsigned)((srow * LDH + scol) * 2);

  v8f acc[2][4];
  v8f zero = {};
#pragma unroll
  for (int i = 0; i < 2; ++i)
#pragma unroll
    for (int j = 0; j < 4; ++j) acc[i][j] = zero;

  v4f f0, f1, f2, f3;  // in-flight X chunk (fp32 path)

  // ---- prologue: stage chunk 0 into buffer 0 ----
#if HAVE_TDM
  if (w == 0) {
    tdm_load_2d((unsigned)(size_t)&Ws[0][0],
                (unsigned long long)(size_t)(Wh + (size_t)n0 * K),
                TK, TN, K, N, K, 3, 3);
    if constexpr (XF16)
      tdm_load_2d((unsigned)(size_t)&Xs[0][0],
                  (unsigned long long)(size_t)(Xh + (size_t)m0 * K),
                  TK, TM, K, M, K, 3, 3);
  }
#else
  async_copy32((unsigned)(size_t)&Ws[0][0] + ldsoff,
               (unsigned long long)(size_t)(Wh + wgoff));
  if constexpr (XF16)
    async_copy32((unsigned)(size_t)&Xs[0][0] + ldsoff,
                 (unsigned long long)(size_t)(Xh + xgoff));
#endif
  if constexpr (!XF16) {
    const float* px = Xf + xgoff;
    f0 = *(const v4f*)px;       f1 = *(const v4f*)(px + 4);
    f2 = *(const v4f*)(px + 8); f3 = *(const v4f*)(px + 12);
    v8h h0, h1;
#pragma unroll
    for (int i = 0; i < 4; ++i) {
      h0[i] = (_Float16)f0[i]; h0[i + 4] = (_Float16)f1[i];
      h1[i] = (_Float16)f2[i]; h1[i + 4] = (_Float16)f3[i];
    }
    *(v8h*)&Xs[0][srow * LDH + scol]     = h0;
    *(v8h*)&Xs[0][srow * LDH + scol + 8] = h1;
  }

  for (int i = 0; i < NIT; ++i) {
    const int cur = i & 1, nxt = cur ^ 1;
    const bool havenext = (i + 1) < NIT;
    if (havenext) {
      const int kn = (i + 1) * TK;
#if HAVE_TDM
      if (w == 0) {
        tdm_load_2d((unsigned)(size_t)&Ws[nxt][0],
                    (unsigned long long)(size_t)(Wh + (size_t)n0 * K + kn),
                    TK, TN, K, N, K, 3, 3);
        if constexpr (XF16)
          tdm_load_2d((unsigned)(size_t)&Xs[nxt][0],
                      (unsigned long long)(size_t)(Xh + (size_t)m0 * K + kn),
                      TK, TM, K, M, K, 3, 3);
      }
      if constexpr (XF16) __builtin_amdgcn_s_wait_tensorcnt((short)2);
      else                __builtin_amdgcn_s_wait_tensorcnt((short)1);
#else
      async_copy32((unsigned)(size_t)&Ws[nxt][0] + ldsoff,
                   (unsigned long long)(size_t)(Wh + wgoff + kn));
      if constexpr (XF16)
        async_copy32((unsigned)(size_t)&Xs[nxt][0] + ldsoff,
                     (unsigned long long)(size_t)(Xh + xgoff + kn));
      if constexpr (XF16) asm volatile("s_wait_asynccnt 0x4" ::: "memory");
      else                asm volatile("s_wait_asynccnt 0x2" ::: "memory");
#endif
      if constexpr (!XF16) {
        const float* px = Xf + xgoff + kn;
        f0 = *(const v4f*)px;       f1 = *(const v4f*)(px + 4);
        f2 = *(const v4f*)(px + 8); f3 = *(const v4f*)(px + 12);
      }
    } else {
#if HAVE_TDM
      __builtin_amdgcn_s_wait_tensorcnt((short)0);
#else
      asm volatile("s_wait_asynccnt 0x0" ::: "memory");
#endif
    }
    __syncthreads();

    // ---- WMMA from buffer `cur` ----
    {
      const int half = lane >> 4;
      const int l16  = lane & 15;
      v16h af[2], bf[4];
#pragma unroll
      for (int mt = 0; mt < 2; ++mt) {
        const _Float16* base = &Xs[cur][(wm * 32 + mt * 16 + l16) * LDH];
        v8h lo = *(const v8h*)(base + half * 8);            // A: k=(e/8)*16+half*8+e%8
        v8h hi = *(const v8h*)(base + 16 + half * 8);
        af[mt] = cat16(lo, hi);
      }
#pragma unroll
      for (int nt = 0; nt < 4; ++nt) {
        const _Float16* base = &Ws[cur][(wn * 64 + nt * 16 + l16) * LDH];
        v8h lo = *(const v8h*)(base + half * 8);            // B: k=(e/8)*16+half*8+e%8
        v8h hi = *(const v8h*)(base + 16 + half * 8);
        bf[nt] = cat16(lo, hi);
      }
#pragma unroll
      for (int mt = 0; mt < 2; ++mt)
#pragma unroll
        for (int nt = 0; nt < 4; ++nt)
          acc[mt][nt] = __builtin_amdgcn_wmma_f32_16x16x32_f16(
              false, af[mt], false, bf[nt], (short)0, acc[mt][nt], false, false);
    }

    if constexpr (!XF16) {
      if (havenext) {
        v8h h0, h1;
#pragma unroll
        for (int j = 0; j < 4; ++j) {
          h0[j] = (_Float16)f0[j]; h0[j + 4] = (_Float16)f1[j];
          h1[j] = (_Float16)f2[j]; h1[j + 4] = (_Float16)f3[j];
        }
        *(v8h*)&Xs[nxt][srow * LDH + scol]     = h0;
        *(v8h*)&Xs[nxt][srow * LDH + scol + 8] = h1;
      }
    }
    __syncthreads();
  }

  // ---- epilogue ----
  const int half = lane >> 4;
  const int l16  = lane & 15;
  if constexpr (MODE == 2) {
    float* Y = (float*)Yv;
#pragma unroll
    for (int mt = 0; mt < 2; ++mt)
#pragma unroll
      for (int nt = 0; nt < 4; ++nt) {
        int n = n0 + wn * 64 + nt * 16 + l16;
        float bn = bias[n];
#pragma unroll
        for (int r = 0; r < 8; ++r) {
          int m = m0 + wm * 32 + mt * 16 + r + half * 8;
          Y[(size_t)m * N + n] = (acc[mt][nt][r] + bn) * scale;
        }
      }
  } else {
    _Float16* Y = (_Float16*)Yv;
#pragma unroll
    for (int mt = 0; mt < 2; ++mt)
#pragma unroll
      for (int nt = 0; nt < 4; ++nt) {
        int n = n0 + wn * 64 + nt * 16 + l16;
        float bn = bias[n];
        int hh = n >> 7, dd = n & 127;
#pragma unroll
        for (int r = 0; r < 8; ++r) {
          int m  = m0 + wm * 32 + mt * 16 + r + half * 8;
          int bb = m >> 8, ss = m & 255;
          float val = (acc[mt][nt][r] + bn) * scale;
          size_t idx = (MODE == 0)
              ? ((((size_t)(bb * NHEAD + hh)) * SEQ + ss) * HDIM + dd)
              : ((((size_t)(bb * NHEAD + hh)) * HDIM + dd) * SEQ + ss);
          Y[idx] = (_Float16)val;
        }
      }
  }
}

// ---------------------------------------------------------------------------
__global__ __launch_bounds__(256)
void cvt_f32_f16(const float* __restrict__ in, _Float16* __restrict__ out, int n8)
{
  int i = blockIdx.x * 256 + threadIdx.x;
  if (i < n8) {
    v4f a = *(const v4f*)(in + (size_t)i * 8);
    v4f b = *(const v4f*)(in + (size_t)i * 8 + 4);
    v8h h;
#pragma unroll
    for (int j = 0; j < 4; ++j) { h[j] = (_Float16)a[j]; h[j + 4] = (_Float16)b[j]; }
    *(v8h*)(out + (size_t)i * 8) = h;
  }
}

// ---------------------------------------------------------------------------
// Fused flash attention. Block = 8 waves sharing one (b,h); K/V chunks staged
// cooperatively into LDS with double-buffered async copies.  Fragments are
// preloaded into distinct register arrays so LDS latency overlaps the WMMA
// chain and the softmax VALU work.
// qh,kh: (B,H,S,HD) f16;  vt: (B,H,HD,S) f16;  out: (B,S,D) f16
// ---------------------------------------------------------------------------
__global__ __launch_bounds__(256)
void attn_fused(const _Float16* __restrict__ qh, const _Float16* __restrict__ kh,
                const _Float16* __restrict__ vt, const int* __restrict__ amask,
                const float* __restrict__ bias_table, _Float16* __restrict__ out)
{
  __shared__ __attribute__((aligned(16))) _Float16 Kl[2][32 * KLDH];   // 32 x 128 chunk
  __shared__ __attribute__((aligned(16))) _Float16 Vl[2][128 * LDH];   // 128 x 32 chunk
  __shared__ __attribute__((aligned(16))) _Float16 Pl[8 * 16 * LDH];

  const int tid  = threadIdx.x;
  const int lane = tid & 31;
  const int w    = tid >> 5;
  const int half = lane >> 4;
  const int l16  = lane & 15;

  const int bid = blockIdx.x;
  const int b   = bid / (NHEAD * 2);
  const int rem = bid % (NHEAD * 2);
  const int h   = rem >> 1;
  const int sq0 = (rem & 1) * 128 + w * 16;

  const size_t bh = (size_t)(b * NHEAD + h);
  const _Float16* Qb = qh + bh * SEQ * HDIM;
  const _Float16* Kb = kh + bh * SEQ * HDIM;
  const _Float16* Vb = vt + bh * HDIM * SEQ;
  const float*    Bt = bias_table + (size_t)h * SEQ * SEQ;
  _Float16*       Pw = &Pl[w * 16 * LDH];

  // staging assignments (256 threads x 32B per tile)
  const int krow = tid >> 3, kcol = (tid & 7) * 16;   // K: 32 rows x 256B
  const int vrow = tid >> 1, vcol = (tid & 1) * 16;   // V: 128 rows x 64B

  auto stageKV = [&](int buf, int sk) {
    async_copy32((unsigned)(size_t)&Kl[buf][krow * KLDH + kcol],
                 (unsigned long long)(size_t)(Kb + (size_t)(sk + krow) * HDIM + kcol));
    async_copy32((unsigned)(size_t)&Vl[buf][vrow * LDH + vcol],
                 (unsigned long long)(size_t)(Vb + (size_t)vrow * SEQ + sk + vcol));
  };

  v16h qf[4];
#pragma unroll
  for (int c = 0; c < 4; ++c) {
    const _Float16* base = Qb + (size_t)(sq0 + l16) * HDIM + c * 32;
    v8h lo = *(const v8h*)(base + half * 8);
    v8h hi = *(const v8h*)(base + 16 + half * 8);
    qf[c] = cat16(lo, hi);
  }

  v8f o[8];
  v8f zero = {};
#pragma unroll
  for (int i = 0; i < 8; ++i) o[i] = zero;
  float mrow[8], lrow[8];
#pragma unroll
  for (int r = 0; r < 8; ++r) { mrow[r] = -3.0e38f; lrow[r] = 0.0f; }

  stageKV(0, 0);

  for (int kc = 0; kc < 8; ++kc) {
    const int cur = kc & 1, nxt = cur ^ 1;
    const bool havenext = (kc + 1) < 8;
    if (havenext) {
      stageKV(nxt, (kc + 1) * 32);
      asm volatile("s_wait_asynccnt 0x2" ::: "memory");
    } else {
      asm volatile("s_wait_asynccnt 0x0" ::: "memory");
    }
    __syncthreads();

    const int sk0 = kc * 32;

    // preload all 8 K fragments (distinct regs -> batched ds_loads, incremental waits)
    v16h kf[2][4];
#pragma unroll
    for (int t = 0; t < 2; ++t)
#pragma unroll
      for (int c = 0; c < 4; ++c) {
        const _Float16* base = &Kl[cur][(t * 16 + l16) * KLDH + c * 32 + half * 8];
        kf[t][c] = cat16(*(const v8h*)base, *(const v8h*)(base + 16));
      }
    int mk0 = amask[b * SEQ + sk0 + l16];
    int mk1 = amask[b * SEQ + sk0 + 16 + l16];

    v8f sc[2];
#pragma unroll
    for (int t = 0; t < 2; ++t) {
      v8f s = zero;
#pragma unroll
      for (int c = 0; c < 4; ++c)
        s = __builtin_amdgcn_wmma_f32_16x16x32_f16(
            false, qf[c], false, kf[t][c], (short)0, s, false, false);
      sc[t] = s;
    }

    // preload V fragments now: LDS latency hides under the softmax below
    v16h vf[8];
#pragma unroll
    for (int nt = 0; nt < 8; ++nt) {
      const _Float16* vb = &Vl[cur][(nt * 16 + l16) * LDH + half * 8];
      vf[nt] = cat16(*(const v8h*)vb, *(const v8h*)(vb + 16));
    }

    // bias + mask (masked_fill after bias, like the reference)
#pragma unroll
    for (int t = 0; t < 2; ++t) {
      int mk = t ? mk1 : mk0;
#pragma unroll
      for (int r = 0; r < 8; ++r) {
        int m = sq0 + r + half * 8;
        float v = sc[t][r] + Bt[(size_t)m * SEQ + sk0 + t * 16 + l16];
        sc[t][r] = (mk == 0) ? -3.0e38f : v;
      }
    }

    // online softmax update, rows live across 16 lanes of each half
#pragma unroll
    for (int r = 0; r < 8; ++r) {
      float tm = fmaxf(sc[0][r], sc[1][r]);
      tm = fmaxf(tm, __shfl_xor(tm, 1, 32));
      tm = fmaxf(tm, __shfl_xor(tm, 2, 32));
      tm = fmaxf(tm, __shfl_xor(tm, 4, 32));
      tm = fmaxf(tm, __shfl_xor(tm, 8, 32));
      float mnew  = fmaxf(mrow[r], tm);
      float alpha = __expf(mrow[r] - mnew);
      float p0 = __expf(sc[0][r] - mnew);
      float p1 = __expf(sc[1][r] - mnew);
      float rs = p0 + p1;
      rs += __shfl_xor(rs, 1, 32);
      rs += __shfl_xor(rs, 2, 32);
      rs += __shfl_xor(rs, 4, 32);
      rs += __shfl_xor(rs, 8, 32);
      lrow[r] = lrow[r] * alpha + rs;
      mrow[r] = mnew;
#pragma unroll
      for (int nt = 0; nt < 8; ++nt) o[nt][r] *= alpha;
      int row = r + half * 8;
      Pw[row * LDH + l16]      = (_Float16)p0;
      Pw[row * LDH + 16 + l16] = (_Float16)p1;
    }

    // reload P in A-operand layout, then PV
    {
      const _Float16* base = &Pw[l16 * LDH];
      v8h lo = *(const v8h*)(base + half * 8);
      v8h hi = *(const v8h*)(base + 16 + half * 8);
      v16h pa = cat16(lo, hi);
#pragma unroll
      for (int nt = 0; nt < 8; ++nt)
        o[nt] = __builtin_amdgcn_wmma_f32_16x16x32_f16(
            false, pa, false, vf[nt], (short)0, o[nt], false, false);
    }
    __syncthreads();
  }

#pragma unroll
  for (int r = 0; r < 8; ++r) {
    float inv = 1.0f / lrow[r];
    int m = sq0 + r + half * 8;
    _Float16* od = out + ((size_t)b * SEQ + m) * DMODEL + h * HDIM;
#pragma unroll
    for (int nt = 0; nt < 8; ++nt)
      od[nt * 16 + l16] = (_Float16)(o[nt][r] * inv);
  }
}

// ---------------------------------------------------------------------------
extern "C" void kernel_launch(void* const* d_in, const int* in_sizes, int n_in,
                              void* d_out, int out_size, void* d_ws, size_t ws_size,
                              hipStream_t stream)
{
  const float* q     = (const float*)d_in[0];
  const float* k     = (const float*)d_in[1];
  const float* v     = (const float*)d_in[2];
  const int*   amask = (const int*)d_in[3];
  const float* Wq    = (const float*)d_in[4];
  const float* bq    = (const float*)d_in[5];
  const float* Wk    = (const float*)d_in[6];
  const float* bk    = (const float*)d_in[7];
  const float* Wv    = (const float*)d_in[8];
  const float* bv    = (const float*)d_in[9];
  const float* Wo    = (const float*)d_in[10];
  const float* bo    = (const float*)d_in[11];
  const float* bt    = (const float*)d_in[12];

  const size_t NE = (size_t)BATCH * NHEAD * SEQ * HDIM;  // 33.55M
  const size_t WN = (size_t)DMODEL * DMODEL;             // 4.19M
  _Float16* qh  = (_Float16*)d_ws;
  _Float16* kh  = qh + NE;
  _Float16* vt  = kh + NE;
  _Float16* ao  = vt + NE;
  _Float16* Whq = ao + NE;
  _Float16* Whk = Whq + WN;
  _Float16* Whv = Whk + WN;
  _Float16* Who = Whv + WN;

  dim3 blk(256);
  dim3 cgrid((unsigned)(WN / 8 / 256));
  cvt_f32_f16<<<cgrid, blk, 0, stream>>>(Wq, Whq, (int)(WN / 8));
  cvt_f32_f16<<<cgrid, blk, 0, stream>>>(Wk, Whk, (int)(WN / 8));
  cvt_f32_f16<<<cgrid, blk, 0, stream>>>(Wv, Whv, (int)(WN / 8));
  cvt_f32_f16<<<cgrid, blk, 0, stream>>>(Wo, Who, (int)(WN / 8));

  dim3 grid(16384 / TM, DMODEL / TN);                    // (128, 16)
  const float qscale = 0.08838834764831845f;             // 1/sqrt(128)

  gemm_xwT<0, 0><<<grid, blk, 0, stream>>>((const void*)q, Whq, bq, (void*)qh, qscale);
  gemm_xwT<0, 0><<<grid, blk, 0, stream>>>((const void*)k, Whk, bk, (void*)kh, 1.0f);
  gemm_xwT<0, 1><<<grid, blk, 0, stream>>>((const void*)v, Whv, bv, (void*)vt, 1.0f);

  attn_fused<<<dim3(BATCH * NHEAD * 2), blk, 0, stream>>>(qh, kh, vt, amask, bt, ao);

  gemm_xwT<1, 2><<<grid, blk, 0, stream>>>((const void*)ao, Who, bo, d_out, 1.0f);
}
